// DDSCTransformer_18545668784248
// MI455X (gfx1250) — matrix operand order repaired and checked by reading, not compiled, not run on hardware
//
#include <hip/hip_runtime.h>
#include <hip/hip_bf16.h>
#include <math.h>

// ---------------------------------------------------------------------------
// DDSCTransformer forward on gfx1250 (MI455X).
// All contractions run through v_wmma_f32_16x16x32_bf16 (f32 accumulate).
// Attention is flash-style: scores stay in registers/LDS, never hit HBM.
// V tiles staged with GLOBAL_LOAD_ASYNC_TO_LDS_B64 (ASYNCcnt-tracked).
// ---------------------------------------------------------------------------

typedef __bf16 bf16;
typedef __attribute__((ext_vector_type(16))) __bf16 bf16x16;
typedef __attribute__((ext_vector_type(8)))  float  floatx8;
typedef int v2i __attribute__((vector_size(2 * sizeof(int))));

#define B_   8
#define CIN_ 6
#define C_   64
#define L_   1024
#define H_   8
#define CH_  512   // C_*H_
#define FF_  256   // 4*C_

#define WMMA_BF16(A, Bm, Cacc) \
  __builtin_amdgcn_wmma_f32_16x16x32_bf16(false, (A), false, (Bm), (short)0, (Cacc), false, false)

#ifndef __has_builtin
#define __has_builtin(x) 0
#endif
#if __has_builtin(__builtin_amdgcn_global_load_async_to_lds_b64)
#define HAS_ASYNC 1
#else
#define HAS_ASYNC 0
#endif

#define AS1 __attribute__((address_space(1)))
#define AS3 __attribute__((address_space(3)))

// Copy 8 bytes global -> LDS. Async (ASYNCcnt) on gfx1250; the builtin's
// existence on this toolchain is probe-confirmed (typed v2i* pointers).
__device__ __forceinline__ void g2l_b64(const void* g, void* l) {
#if HAS_ASYNC
  __builtin_amdgcn_global_load_async_to_lds_b64(
      (AS1 v2i*)(uintptr_t)g,
      (AS3 v2i*)(unsigned)(uintptr_t)l, 0, 0);
#else
  *(unsigned long long*)l = *(const unsigned long long*)g;
#endif
}
__device__ __forceinline__ void g2l_wait() {
#if HAS_ASYNC
  asm volatile("s_wait_asynccnt 0x0" ::: "memory");
#endif
}

union pk4 { bf16 h[4]; unsigned long long u; };
union pk2 { bf16 h[2]; unsigned u; };

// --- WMMA fragment loaders (layouts per CDNA5 ISA §7.12.2) -------
// A (16x32 bf16, MxK): lane lo -> M; VGPR j holds K pair:
//   K = (j>=4?16:0) + 8*hi + 2*(j&3)   (hi = lane>>4)
// B (32x16 bf16, KxN): lane lo -> N; VGPR j holds K pair: same K as A
// LDS tiles are stored [row][K] with even stride so K-pairs are one b32 read.
__device__ __forceinline__ bf16x16 frag_a(const bf16* lds, int row_base,
                                          int stride, int kbase, int lane) {
  int lo = lane & 15, hi = lane >> 4;
  union { bf16x16 v; unsigned u[8]; } r;
  const bf16* rowp = lds + (size_t)(row_base + lo) * stride;
#pragma unroll
  for (int j = 0; j < 8; ++j) {
    int k = kbase + ((j & 4) ? 16 : 0) + hi * 8 + (j & 3) * 2;
    r.u[j] = *(const unsigned*)(rowp + k);
  }
  return r.v;
}

__device__ __forceinline__ bf16x16 frag_b(const bf16* lds, int row_base,
                                          int stride, int kbase, int lane) {
  int lo = lane & 15, hi = lane >> 4;
  union { bf16x16 v; unsigned u[8]; } r;
  const bf16* rowp = lds + (size_t)(row_base + lo) * stride;
#pragma unroll
  for (int j = 0; j < 8; ++j) {
    int k = kbase + ((j & 4) ? 16 : 0) + hi * 8 + (j & 3) * 2;
    r.u[j] = *(const unsigned*)(rowp + k);
  }
  return r.v;
}

// --- Generic pointwise-conv GEMM: Out[b,o,l] = sum_c W[o,c]*In[b,c,l] -------
// grid = (L/128, O/64, B), block = 256 (8 waves). Workgroup tile 64o x 128l;
// wave w owns l-subtile w and all four 16-row o-subtiles (4 WMMA / K-chunk).
__global__ __launch_bounds__(256) void k_gemm(const float* __restrict__ W,
                                              const float* __restrict__ In,
                                              float* __restrict__ Out,
                                              int O, int Cin, int relu) {
  __shared__ __align__(16) bf16 Alds[64 * 40];
  __shared__ __align__(16) bf16 Blds[128 * 40];
  const int tid = threadIdx.x, lane = tid & 31, w = tid >> 5;
  const int lb = blockIdx.x * 128, ob = blockIdx.y * 64, b = blockIdx.z;
  const float* inb = In + (size_t)b * Cin * L_;
  floatx8 acc0 = (floatx8)0.0f, acc1 = acc0, acc2 = acc0, acc3 = acc0;

  for (int kk = 0; kk < Cin; kk += 32) {
    // stage A: 64x32 f32 -> bf16; float4 loads, b64 LDS stores (2 per thread)
#pragma unroll
    for (int r = 0; r < 2; ++r) {
      int idx = tid + r * 256;
      int m = idx >> 3, k4 = (idx & 7) * 4;
      const float4 wv = *(const float4*)&W[(size_t)(ob + m) * Cin + kk + k4];
      pk4 p; p.h[0] = (bf16)wv.x; p.h[1] = (bf16)wv.y;
             p.h[2] = (bf16)wv.z; p.h[3] = (bf16)wv.w;
      *(unsigned long long*)&Alds[m * 40 + k4] = p.u;
    }
    // stage B transposed [n][k]: 4 coalesced b32 loads -> one b64 LDS store
#pragma unroll
    for (int r = 0; r < 4; ++r) {
      int idx = tid + r * 256;
      int n = idx & 127, k0 = (idx >> 7) * 4;
      pk4 p;
      p.h[0] = (bf16)inb[(size_t)(kk + k0 + 0) * L_ + lb + n];
      p.h[1] = (bf16)inb[(size_t)(kk + k0 + 1) * L_ + lb + n];
      p.h[2] = (bf16)inb[(size_t)(kk + k0 + 2) * L_ + lb + n];
      p.h[3] = (bf16)inb[(size_t)(kk + k0 + 3) * L_ + lb + n];
      *(unsigned long long*)&Blds[n * 40 + k0] = p.u;
    }
    __syncthreads();
    const bf16x16 bm = frag_b(Blds, w * 16, 40, 0, lane);
    acc0 = WMMA_BF16(frag_a(Alds,  0, 40, 0, lane), bm, acc0);
    acc1 = WMMA_BF16(frag_a(Alds, 16, 40, 0, lane), bm, acc1);
    acc2 = WMMA_BF16(frag_a(Alds, 32, 40, 0, lane), bm, acc2);
    acc3 = WMMA_BF16(frag_a(Alds, 48, 40, 0, lane), bm, acc3);
    __syncthreads();
  }

  const int lo = lane & 15, hi = lane >> 4;
  float* outp = Out + (size_t)b * O * L_ + lb + w * 16 + lo;
#pragma unroll
  for (int i = 0; i < 8; ++i) {
    int orow = ob + i + 8 * hi;
    float v0 = acc0[i], v1 = acc1[i], v2 = acc2[i], v3 = acc3[i];
    if (relu) {
      v0 = fmaxf(v0, 0.0f); v1 = fmaxf(v1, 0.0f);
      v2 = fmaxf(v2, 0.0f); v3 = fmaxf(v3, 0.0f);
    }
    outp[(size_t)(orow)      * L_] = v0;
    outp[(size_t)(orow + 16) * L_] = v1;
    outp[(size_t)(orow + 32) * L_] = v2;
    outp[(size_t)(orow + 48) * L_] = v3;
  }
}

// --- Flash attention per (b, h, 128-query tile) -----------------------------
// Q/K pre-scaled by c^-0.25.  S = K^T Q, softmax over keys, O = V P.
__global__ __launch_bounds__(256) void k_attn(const bf16* __restrict__ Q,
                                              const bf16* __restrict__ K,
                                              const bf16* __restrict__ V,
                                              float* __restrict__ Oh) {
  __shared__ __align__(16) bf16 qlds[128 * 72];   // [q][c]
  __shared__ __align__(16) bf16 klds[32 * 72];    // [k][c]
  __shared__ __align__(16) bf16 vlds[64 * 40];    // [c][k]
  __shared__ __align__(16) bf16 plds[8][16 * 40]; // per-wave [q][k]
  const int tid = threadIdx.x, lane = tid & 31, w = tid >> 5;
  const int lo = lane & 15, hi = lane >> 4;
  const int q0 = blockIdx.x * 128, h = blockIdx.y, b = blockIdx.z;
  const size_t base = ((size_t)b * CH_ + h * C_) * L_;

  // stage Q transposed [q][c]; b32 global loads carry two q's
#pragma unroll
  for (int r = 0; r < 16; ++r) {
    int idx = tid + r * 256;                 // 4096 pair-groups
    int q2 = (idx & 63) * 2, c = idx >> 6;
    pk2 p; p.u = *(const unsigned*)&Q[base + (size_t)c * L_ + q0 + q2];
    qlds[(q2 + 0) * 72 + c] = p.h[0];
    qlds[(q2 + 1) * 72 + c] = p.h[1];
  }
  __syncthreads();
  const bf16x16 bq0 = frag_b(qlds, w * 16, 72, 0,  lane);
  const bf16x16 bq1 = frag_b(qlds, w * 16, 72, 32, lane);

  floatx8 O0 = (floatx8)0.0f, O1 = O0, O2 = O0, O3 = O0;
  float m_run = -INFINITY, l_run = 0.0f;

  for (int kc = 0; kc < L_; kc += 32) {
    __syncthreads();                                     // guard LDS re-stage
    if (kc + 32 < L_) {                                  // global_prefetch_b8
      __builtin_prefetch(K + base + (size_t)(tid & 63) * L_ + kc + 32, 0, 1);
    }
    // K chunk transposed [k][c]; b32 loads carry two k's (4 groups / thread)
#pragma unroll
    for (int r = 0; r < 4; ++r) {
      int idx = tid + r * 256;               // 1024 pair-groups (64c x 16)
      int k2 = (idx & 15) * 2, c = idx >> 4;
      pk2 p; p.u = *(const unsigned*)&K[base + (size_t)c * L_ + kc + k2];
      klds[(k2 + 0) * 72 + c] = p.h[0];
      klds[(k2 + 1) * 72 + c] = p.h[1];
    }
    // V chunk natural [c][k]; async global->LDS b64 (2 transfers / thread)
#pragma unroll
    for (int r = 0; r < 2; ++r) {
      int idx = tid + r * 256;               // 512 b64 transfers (64c x 8)
      int k4 = (idx & 7) * 4, c = idx >> 3;
      g2l_b64(&V[base + (size_t)c * L_ + kc + k4], &vlds[c * 40 + k4]);
    }
    g2l_wait();
    __syncthreads();

    bf16x16 ak00 = frag_a(klds, 0, 72, 0, lane);
    bf16x16 ak01 = frag_a(klds, 0, 72, 32, lane);
    bf16x16 ak10 = frag_a(klds, 16, 72, 0, lane);
    bf16x16 ak11 = frag_a(klds, 16, 72, 32, lane);
    floatx8 Sa = (floatx8)0.0f, Sb = (floatx8)0.0f;
    Sa = WMMA_BF16(ak00, bq0, Sa);  Sa = WMMA_BF16(ak01, bq1, Sa);
    Sb = WMMA_BF16(ak10, bq0, Sb);  Sb = WMMA_BF16(ak11, bq1, Sb);

    // ---- online softmax over keys; query = lane column (lo), replicated halves
    float cm = -INFINITY;
#pragma unroll
    for (int i = 0; i < 8; ++i) { cm = fmaxf(cm, Sa[i]); cm = fmaxf(cm, Sb[i]); }
    cm = fmaxf(cm, __shfl_xor(cm, 16));
    const float mn   = fmaxf(m_run, cm);
    const float corr = __expf(m_run - mn);
    float rsum = 0.0f;
    bf16* pw_ = plds[w];
#pragma unroll
    for (int i = 0; i < 8; ++i) {
      float pa = __expf(Sa[i] - mn); rsum += pa;
      pw_[lo * 40 + (i + 8 * hi)]      = (bf16)pa;       // keys 0..15
      float pb = __expf(Sb[i] - mn); rsum += pb;
      pw_[lo * 40 + (16 + i + 8 * hi)] = (bf16)pb;       // keys 16..31
    }
    rsum += __shfl_xor(rsum, 16);
    l_run = l_run * corr + rsum;
    m_run = mn;
    O0 *= corr; O1 *= corr; O2 *= corr; O3 *= corr;

    bf16x16 bp = frag_b(pw_, 0, 40, 0, lane);            // P as B operand
    bf16x16 av;
    av = frag_a(vlds,  0, 40, 0, lane);  O0 = WMMA_BF16(av, bp, O0);
    av = frag_a(vlds, 16, 40, 0, lane);  O1 = WMMA_BF16(av, bp, O1);
    av = frag_a(vlds, 32, 40, 0, lane);  O2 = WMMA_BF16(av, bp, O2);
    av = frag_a(vlds, 48, 40, 0, lane);  O3 = WMMA_BF16(av, bp, O3);
  }

  const float inv = 1.0f / l_run;
  const size_t ob = base + q0 + w * 16 + lo;
#pragma unroll
  for (int i = 0; i < 8; ++i) {
    int cr = i + 8 * hi;
    Oh[ob + (size_t)(cr)      * L_] = O0[i] * inv;
    Oh[ob + (size_t)(cr + 16) * L_] = O1[i] * inv;
    Oh[ob + (size_t)(cr + 32) * L_] = O2[i] * inv;
    Oh[ob + (size_t)(cr + 48) * L_] = O3[i] * inv;
  }
}

// --- Encoder pointwise 6 -> 64 ---------------------------------------------
__global__ __launch_bounds__(256) void k_enc_pw(const float* __restrict__ x,
                                                const float* __restrict__ pw,
                                                float* __restrict__ out) {
  int t = blockIdx.x * 256 + threadIdx.x;                // B*64*1024
  if (t >= B_ * C_ * L_) return;
  int l = t & (L_ - 1), oc = (t >> 10) & 63, b = t >> 16;
  float s = 0.0f;
#pragma unroll
  for (int c = 0; c < CIN_; ++c)
    s += pw[oc * CIN_ + c] * x[((size_t)b * CIN_ + c) * L_ + l];
  out[t] = s;
}

// --- Gated depthwise conv (k=3 and k=15, softmax gate) ----------------------
__global__ __launch_bounds__(256) void k_dds_dw(const float* __restrict__ in,
                                                const float* __restrict__ w3,
                                                const float* __restrict__ w15,
                                                const float* __restrict__ gate,
                                                int Cout, float* __restrict__ outF,
                                                bf16* __restrict__ outB, float scale) {
  int t = blockIdx.x * 256 + threadIdx.x;
  if (t >= B_ * Cout * L_) return;
  int l = t & (L_ - 1), oc = (t >> 10) % Cout;
  const float* row = in + (size_t)(t >> 10) * L_;
  float e0 = __expf(gate[0]), e1 = __expf(gate[1]);
  float g0 = e0 / (e0 + e1), g1 = 1.0f - g0;
  float a3 = 0.0f;
#pragma unroll
  for (int j = 0; j < 3; ++j) {
    int li = l + j - 1;
    if (li >= 0 && li < L_) a3 += row[li] * w3[oc * 3 + j];
  }
  float a15 = 0.0f;
#pragma unroll
  for (int j = 0; j < 15; ++j) {
    int li = l + j - 7;
    if (li >= 0 && li < L_) a15 += row[li] * w15[oc * 15 + j];
  }
  float y = (g0 * a3 + g1 * a15) * scale;
  if (outF) outF[t] = y;
  if (outB) outB[t] = (bf16)y;
}

// --- Fused residual add + InstanceNorm1d (one block per (b,c) row) ----------
__global__ __launch_bounds__(256) void k_inorm_add(const float* __restrict__ a,
                                                   const float* __restrict__ b,
                                                   const float* __restrict__ g,
                                                   const float* __restrict__ be,
                                                   float* __restrict__ out) {
  __shared__ float r1[256], r2[256];
  int row = blockIdx.x, tid = threadIdx.x, c = row & (C_ - 1);
  const float* pa = a + (size_t)row * L_;
  const float* pb = b + (size_t)row * L_;
  float v[4], s1 = 0.0f, s2 = 0.0f;
#pragma unroll
  for (int i = 0; i < 4; ++i) {
    float t = pa[tid + i * 256] + pb[tid + i * 256];
    v[i] = t; s1 += t; s2 += t * t;
  }
  r1[tid] = s1; r2[tid] = s2; __syncthreads();
  for (int o = 128; o > 0; o >>= 1) {
    if (tid < o) { r1[tid] += r1[tid + o]; r2[tid] += r2[tid + o]; }
    __syncthreads();
  }
  float mu = r1[0] * (1.0f / L_);
  float var = r2[0] * (1.0f / L_) - mu * mu;
  float rs = rsqrtf(var + 1e-5f);
  float gg = g[c], bb = be[c];
  float* po = out + (size_t)row * L_;
#pragma unroll
  for (int i = 0; i < 4; ++i) po[tid + i * 256] = (v[i] - mu) * rs * gg + bb;
}

// --- Output head: 64 -> 1 + bias + sigmoid ----------------------------------
__global__ __launch_bounds__(256) void k_head(const float* __restrict__ act,
                                              const float* __restrict__ wo,
                                              const float* __restrict__ bo,
                                              float* __restrict__ out) {
  int t = blockIdx.x * 256 + threadIdx.x;                // B*L
  if (t >= B_ * L_) return;
  int b = t >> 10, l = t & (L_ - 1);
  float s = bo[0];
#pragma unroll 8
  for (int c = 0; c < C_; ++c)
    s += act[((size_t)b * C_ + c) * L_ + l] * wo[c];
  out[t] = 1.0f / (1.0f + __expf(-s));
}

// ---------------------------------------------------------------------------
extern "C" void kernel_launch(void* const* d_in, const int* in_sizes, int n_in,
                              void* d_out, int out_size, void* d_ws, size_t ws_size,
                              hipStream_t stream) {
  (void)in_sizes; (void)n_in; (void)out_size; (void)ws_size;
  int idx = 0;
  auto F = [&](int i) { return (const float*)d_in[i]; };
  struct DDS { const float *pw, *dw3, *dw15, *gate; };
  auto rdDDS = [&]() {
    DDS d; d.pw = F(idx++); d.dw3 = F(idx++); d.dw15 = F(idx++); d.gate = F(idx++);
    return d;
  };
  // setup_inputs() insertion order: x, enc{pw,dw[0],dw[1],gate}, blocks[0..3]
  // each {attn{q,k,v(dds), u}, g1,b1,g2,b2, w1,w2}, wo, bo
  const float* x = F(idx++);
  DDS enc = rdDDS();
  struct Blk { DDS q, k, v; const float *u, *g1, *b1, *g2, *b2, *w1, *w2; } blk[4];
  for (int i = 0; i < 4; ++i) {
    blk[i].q = rdDDS(); blk[i].k = rdDDS(); blk[i].v = rdDDS();
    blk[i].u  = F(idx++);
    blk[i].g1 = F(idx++); blk[i].b1 = F(idx++);
    blk[i].g2 = F(idx++); blk[i].b2 = F(idx++);
    blk[i].w1 = F(idx++); blk[i].w2 = F(idx++);
  }
  const float* wo = F(idx++);
  const float* bo = F(idx++);

  // ---- workspace carve (all scratch lives in d_ws) ----
  char* base = (char*)d_ws; size_t off = 0;
  auto carve = [&](size_t bytes) {
    void* r = base + off; off += (bytes + 255) & ~(size_t)255; return r;
  };
  float* bufA   = (float*)carve((size_t)B_ * C_  * L_ * 4);  // activations ping
  float* bufB   = (float*)carve((size_t)B_ * C_  * L_ * 4);  // activations pong
  float* pw_tmp = (float*)carve((size_t)B_ * CH_ * L_ * 4);  // pointwise scratch
  float* oheads = (float*)carve((size_t)B_ * CH_ * L_ * 4);  // attention heads out
  float* u_out  = (float*)carve((size_t)B_ * C_  * L_ * 4);  // unify / w2 out
  float* ffh    = (float*)carve((size_t)B_ * FF_ * L_ * 4);  // FF hidden
  bf16*  qb     = (bf16*) carve((size_t)B_ * CH_ * L_ * 2);
  bf16*  kb     = (bf16*) carve((size_t)B_ * CH_ * L_ * 2);
  bf16*  vb     = (bf16*) carve((size_t)B_ * CH_ * L_ * 2);

  auto gemm = [&](const float* W, const float* In, float* Out, int O, int Cin, int relu) {
    k_gemm<<<dim3(L_ / 128, O / 64, B_), 256, 0, stream>>>(W, In, Out, O, Cin, relu);
  };
  auto dds512 = [&](const DDS& d, const float* actin, bf16* outB, float scale) {
    gemm(d.pw, actin, pw_tmp, CH_, C_, 0);
    k_dds_dw<<<(B_ * CH_ * L_) / 256, 256, 0, stream>>>(
        pw_tmp, d.dw3, d.dw15, d.gate, CH_, nullptr, outB, scale);
  };

  // encode: pointwise 6->64 then gated depthwise -> bufA
  k_enc_pw<<<(B_ * C_ * L_) / 256, 256, 0, stream>>>(x, enc.pw, u_out);
  k_dds_dw<<<(B_ * C_ * L_) / 256, 256, 0, stream>>>(
      u_out, enc.dw3, enc.dw15, enc.gate, C_, bufA, nullptr, 1.0f);

  const float s = 0.35355339059327373f;                  // 64^-0.25
  for (int i = 0; i < 4; ++i) {
    dds512(blk[i].q, bufA, qb, s);
    dds512(blk[i].k, bufA, kb, s);
    dds512(blk[i].v, bufA, vb, 1.0f);
    k_attn<<<dim3(L_ / 128, H_, B_), 256, 0, stream>>>(qb, kb, vb, oheads);
    gemm(blk[i].u, oheads, u_out, C_, CH_, 0);
    k_inorm_add<<<B_ * C_, 256, 0, stream>>>(u_out, bufA, blk[i].g1, blk[i].b1, bufB);
    gemm(blk[i].w1, bufB, ffh, FF_, C_, 1);
    gemm(blk[i].w2, ffh, u_out, C_, FF_, 0);
    k_inorm_add<<<B_ * C_, 256, 0, stream>>>(u_out, bufB, blk[i].g2, blk[i].b2, bufA);
  }
  k_head<<<(B_ * L_) / 256, 256, 0, stream>>>(bufA, wo, bo, (float*)d_out);
}
